// GNNSATAttention_10514079941113
// MI455X (gfx1250) — matrix operand order repaired and checked by reading, not compiled, not run on hardware
//
#include <hip/hip_runtime.h>
#include <hip/hip_bf16.h>
#include <math.h>

#define H 128
#define STEPS 9

typedef __attribute__((ext_vector_type(16))) __bf16        v16bf;
typedef __attribute__((ext_vector_type(8)))  float         v8f;
typedef __attribute__((ext_vector_type(8)))  unsigned int  v8u;

union FragBF { v8u u; v16bf b; };

__device__ __forceinline__ unsigned short f2bf(float f) {
  unsigned int u = __float_as_uint(f);
  unsigned int r = (u + 0x7FFFu + ((u >> 16) & 1u)) >> 16;
  return (unsigned short)r;
}

__device__ __forceinline__ float sigm(float x) { return 1.0f / (1.0f + expf(-x)); }

// A fragment: 16x32 bf16, row m = lane&15, K interleaved per ISA 7.12.2:
// VGPR i(0..3): K = k0 + 2i + 8*half ; VGPR 4+i: K = k0 + 16 + 2i + 8*half
__device__ __forceinline__ v16bf load_a_frag(const unsigned short* rowp, int k0, int half) {
  FragBF f;
#pragma unroll
  for (int i = 0; i < 4; ++i) {
    int kk = k0 + i * 2 + half * 8;
    f.u[i]     = *(const unsigned int*)(rowp + kk);
    f.u[i + 4] = *(const unsigned int*)(rowp + kk + 16);
  }
  return f.b;
}

// B fragment: 32x16 bf16, col n = lane&15, K interleaved like A:
// VGPR i(0..3): K = k0 + 2i + 8*half ; VGPR 4+i: K = k0 + 16 + 2i + 8*half     (rowp points at W[n][*], row-major [N][K])
__device__ __forceinline__ v16bf load_b_frag(const unsigned short* rowp, int k0, int half) {
  FragBF f;
#pragma unroll
  for (int i = 0; i < 8; ++i) {
    int kk = k0 + half * 8 + (i & 3) * 2 + (i >> 2) * 16;
    f.u[i] = *(const unsigned int*)(rowp + kk);
  }
  return f.b;
}

// Dual-M-panel GEMM: acc0 += act[0:16,:] @ Wtile ; acc1 += act[16:32,:] @ Wtile
// One B fragment feeds two WMMAs -> two independent accumulator chains.
__device__ __forceinline__ void gemm_tile_acc2(const unsigned short* act,
                                               const unsigned short* W,
                                               int n0, int lane, v8f& acc0, v8f& acc1) {
  int l15 = lane & 15, half = lane >> 4;
  const unsigned short* arow0 = act + l15 * H;
  const unsigned short* arow1 = act + (16 + l15) * H;
  const unsigned short* brow  = W + (size_t)(n0 + l15) * H;
#pragma unroll
  for (int k0 = 0; k0 < H; k0 += 32) {
    v16bf b  = load_b_frag(brow, k0, half);
    v16bf a0 = load_a_frag(arow0, k0, half);
    v16bf a1 = load_a_frag(arow1, k0, half);
    acc0 = __builtin_amdgcn_wmma_f32_16x16x32_bf16(false, a0, false, b, (short)0, acc0,
                                                   false, false);
    acc1 = __builtin_amdgcn_wmma_f32_16x16x32_bf16(false, a1, false, b, (short)0, acc1,
                                                   false, false);
  }
}

__device__ __forceinline__ void acc_init(v8f& acc, float v) {
#pragma unroll
  for (int r = 0; r < 8; ++r) acc[r] = v;
}

__device__ __forceinline__ void store_act2_bf16(unsigned short* dst, const v8f& a0,
                                                const v8f& a1, int n0, int lane) {
  int l15 = lane & 15, half = lane >> 4;
#pragma unroll
  for (int r = 0; r < 8; ++r) {
    dst[(r + half * 8) * H + n0 + l15]        = f2bf(a0[r]);
    dst[(16 + r + half * 8) * H + n0 + l15]   = f2bf(a1[r]);
  }
}

// ---------------- fused 3-layer MLP (Linear->ReLU x3) over a 32-row panel -----------
__global__ __launch_bounds__(256) void mlp3_kernel(
    const float* __restrict__ X, int Nrows,
    const unsigned short* __restrict__ W1, const float* __restrict__ b1,
    const unsigned short* __restrict__ W2, const float* __restrict__ b2,
    const unsigned short* __restrict__ W3, const float* __restrict__ b3,
    float* __restrict__ OUT) {
  __shared__ unsigned short actA[32 * H];
  __shared__ unsigned short actB[32 * H];
  int tid = threadIdx.x, lane = tid & 31, wave = tid >> 5;
  int l15 = lane & 15, half = lane >> 4;
  int row0 = blockIdx.x * 32;
  int n0 = wave * 16;

  for (int i = tid; i < 32 * H; i += 256) {
    int m = i >> 7, f = i & (H - 1);
    int rr = row0 + m;
    float v = (rr < Nrows) ? X[(size_t)rr * H + f] : 0.0f;
    actA[i] = f2bf(v);
  }
  __syncthreads();

  v8f acc0, acc1;
  // layer 1
  acc_init(acc0, b1[n0 + l15]);
  acc_init(acc1, b1[n0 + l15]);
  gemm_tile_acc2(actA, W1, n0, lane, acc0, acc1);
#pragma unroll
  for (int r = 0; r < 8; ++r) {
    acc0[r] = fmaxf(acc0[r], 0.0f);
    acc1[r] = fmaxf(acc1[r], 0.0f);
  }
  store_act2_bf16(actB, acc0, acc1, n0, lane);
  __syncthreads();
  // layer 2
  acc_init(acc0, b2[n0 + l15]);
  acc_init(acc1, b2[n0 + l15]);
  gemm_tile_acc2(actB, W2, n0, lane, acc0, acc1);
#pragma unroll
  for (int r = 0; r < 8; ++r) {
    acc0[r] = fmaxf(acc0[r], 0.0f);
    acc1[r] = fmaxf(acc1[r], 0.0f);
  }
  store_act2_bf16(actA, acc0, acc1, n0, lane);
  __syncthreads();
  // layer 3
  acc_init(acc0, b3[n0 + l15]);
  acc_init(acc1, b3[n0 + l15]);
  gemm_tile_acc2(actA, W3, n0, lane, acc0, acc1);
#pragma unroll
  for (int r = 0; r < 8; ++r) {
    float v0 = fmaxf(acc0[r], 0.0f);
    float v1 = fmaxf(acc1[r], 0.0f);
    int rr0 = row0 + r + half * 8;
    int rr1 = rr0 + 16;
    if (rr0 < Nrows) OUT[(size_t)rr0 * H + n0 + l15] = v0;
    if (rr1 < Nrows) OUT[(size_t)rr1 * H + n0 + l15] = v1;
  }
}

// ---------------- LSTM cell: gates = x@Wih.T + h@Whh.T + bih + bhh ----------------
__global__ __launch_bounds__(256) void lstm_kernel(
    const float* __restrict__ X, float* __restrict__ Hs, float* __restrict__ Cs,
    const unsigned short* __restrict__ Wih, const unsigned short* __restrict__ Whh,
    const float* __restrict__ bih, const float* __restrict__ bhh, int Nrows) {
  __shared__ unsigned short xs[32 * H];
  __shared__ unsigned short hs[32 * H];
  __shared__ float gates[32 * 4 * H];  // 32 x 512 fp32 = 64KB
  int tid = threadIdx.x, lane = tid & 31, wave = tid >> 5;
  int l15 = lane & 15, half = lane >> 4;
  int row0 = blockIdx.x * 32;

  for (int i = tid; i < 32 * H; i += 256) {
    int m = i >> 7, f = i & (H - 1);
    int rr = row0 + m;
    float xv = (rr < Nrows) ? X[(size_t)rr * H + f] : 0.0f;
    float hv = (rr < Nrows) ? Hs[(size_t)rr * H + f] : 0.0f;
    xs[i] = f2bf(xv);
    hs[i] = f2bf(hv);
  }
  __syncthreads();

#pragma unroll
  for (int t = 0; t < 4; ++t) {
    int n0 = (wave * 4 + t) * 16;
    float bv = bih[n0 + l15] + bhh[n0 + l15];
    v8f acc0, acc1;
    acc_init(acc0, bv);
    acc_init(acc1, bv);
    gemm_tile_acc2(xs, Wih, n0, lane, acc0, acc1);
    gemm_tile_acc2(hs, Whh, n0, lane, acc0, acc1);
#pragma unroll
    for (int r = 0; r < 8; ++r) {
      gates[(r + half * 8) * 4 * H + n0 + l15]      = acc0[r];
      gates[(16 + r + half * 8) * 4 * H + n0 + l15] = acc1[r];
    }
  }
  __syncthreads();

  for (int i = tid; i < 32 * H; i += 256) {
    int m = i >> 7, f = i & (H - 1);
    int rr = row0 + m;
    if (rr >= Nrows) continue;
    float ig = gates[m * 4 * H + f];
    float fg = gates[m * 4 * H + H + f];
    float gg = gates[m * 4 * H + 2 * H + f];
    float og = gates[m * 4 * H + 3 * H + f];
    float c = Cs[(size_t)rr * H + f];
    c = sigm(fg) * c + sigm(ig) * tanhf(gg);
    float h = sigm(og) * tanhf(c);
    Hs[(size_t)rr * H + f] = h;
    Cs[(size_t)rr * H + f] = fmaxf(c, 0.0f);  // reference: relu AFTER h uses c
  }
}

// ---------------- edge gather + scatter-add (segment sum) ----------------
__global__ __launch_bounds__(256) void scatter_kernel(
    const float* __restrict__ MSG, const int* __restrict__ src,
    const int* __restrict__ dst, int E, float* __restrict__ AGG) {
  int idx = blockIdx.x * 256 + threadIdx.x;
  if (idx >= E * 32) return;
  int e = idx >> 5, q = (idx & 31) * 4;
  int s = src[e], d = dst[e];
  const float4 v = *(const float4*)(MSG + (size_t)s * H + q);
  float* out = AGG + (size_t)d * H + q;
  atomicAdd(out + 0, v.x);
  atomicAdd(out + 1, v.y);
  atomicAdd(out + 2, v.z);
  atomicAdd(out + 3, v.w);
}

// ---------------- init h=c=embed[type] ----------------
__global__ __launch_bounds__(256) void init_nodes_kernel(
    const int* __restrict__ xids, const float* __restrict__ embed,
    float* __restrict__ Hs, float* __restrict__ Cs, int N) {
  long long i = (long long)blockIdx.x * 256 + threadIdx.x;
  if (i >= (long long)N * H) return;
  int node = (int)(i >> 7), f = (int)(i & (H - 1));
  float v = embed[xids[node] * H + f];
  Hs[i] = v;
  Cs[i] = v;
}

__global__ __launch_bounds__(256) void f2bf_kernel(const float* __restrict__ in,
                                                   unsigned short* __restrict__ out, int n) {
  int i = blockIdx.x * 256 + threadIdx.x;
  if (i < n) out[i] = f2bf(in[i]);
}

// ---------------- attention pooling ----------------
__device__ __forceinline__ unsigned int enc_f(float f) {
  int i = __float_as_int(f);
  return (i >= 0) ? ((unsigned int)i | 0x80000000u) : ~(unsigned int)i;
}
__device__ __forceinline__ float dec_f(unsigned int u) {
  int i = (u & 0x80000000u) ? (int)(u & 0x7FFFFFFFu) : (int)~u;
  return __int_as_float(i);
}

__global__ __launch_bounds__(128) void pool_logits_kernel(
    const float* __restrict__ C, const float* __restrict__ gW,
    const float* __restrict__ gb, float* __restrict__ logits,
    unsigned int* __restrict__ maxenc, int N) {
  int j = blockIdx.x, f = threadIdx.x;
  float p = C[(size_t)j * H + f] * gW[f];
#pragma unroll
  for (int off = 16; off > 0; off >>= 1) p += __shfl_down(p, off, 32);
  __shared__ float red[4];
  if ((f & 31) == 0) red[f >> 5] = p;
  __syncthreads();
  if (f == 0) {
    float logit = red[0] + red[1] + red[2] + red[3] + gb[0];
    logits[j] = logit;
    atomicMax(maxenc, enc_f(logit));
  }
}

__global__ __launch_bounds__(128) void pool_accum_kernel(
    const float* __restrict__ C, const float* __restrict__ logits,
    const unsigned int* __restrict__ maxenc, float* __restrict__ acc,
    float* __restrict__ sumexp, int N) {
  int f = threadIdx.x;
  float mx = dec_f(*maxenc);
  float lacc = 0.0f, lsum = 0.0f;
  int j0 = blockIdx.x * 128;
  for (int jj = 0; jj < 128; ++jj) {
    int j = j0 + jj;
    if (j >= N) break;
    float w = expf(logits[j] - mx);
    lacc += w * C[(size_t)j * H + f];
    lsum += w;
  }
  atomicAdd(&acc[f], lacc);
  if (f == 0) atomicAdd(sumexp, lsum);
}

__global__ __launch_bounds__(128) void head_kernel(
    const float* __restrict__ acc, const float* __restrict__ sumexp,
    const float* __restrict__ mW1, const float* __restrict__ mb1,
    const float* __restrict__ mW2, const float* __restrict__ mb2,
    const float* __restrict__ mW3, const float* __restrict__ mb3,
    float* __restrict__ out) {
  __shared__ float x0[H], x1[H], x2[H];
  int f = threadIdx.x;
  x0[f] = acc[f] / (*sumexp);
  __syncthreads();
  float s = mb1[f];
  for (int k = 0; k < H; ++k) s += mW1[f * H + k] * x0[k];
  x1[f] = fmaxf(s, 0.0f);
  __syncthreads();
  s = mb2[f];
  for (int k = 0; k < H; ++k) s += mW2[f * H + k] * x1[k];
  x2[f] = fmaxf(s, 0.0f);
  __syncthreads();
  if (f < 2) {
    s = mb3[f];
    for (int k = 0; k < H; ++k) s += mW3[f * H + k] * x2[k];
    out[f] = s;
  }
}

// ====================================================================================
extern "C" void kernel_launch(void* const* d_in, const int* in_sizes, int n_in,
                              void* d_out, int out_size, void* d_ws, size_t ws_size,
                              hipStream_t stream) {
  const int* var_x    = (const int*)d_in[0];
  const int* clause_x = (const int*)d_in[1];
  const int* pos_src  = (const int*)d_in[2];
  const int* pos_dst  = (const int*)d_in[3];
  const int* neg_src  = (const int*)d_in[4];
  const int* neg_dst  = (const int*)d_in[5];
  const int* posr_src = (const int*)d_in[6];
  const int* posr_dst = (const int*)d_in[7];
  const int* negr_src = (const int*)d_in[8];
  const int* negr_dst = (const int*)d_in[9];
  const float* embed  = (const float*)d_in[10];
  const float* eW1    = (const float*)d_in[11];
  const float* eb1    = (const float*)d_in[12];
  const float* eW2    = (const float*)d_in[13];
  const float* eb2    = (const float*)d_in[14];
  const float* eW3    = (const float*)d_in[15];
  const float* eb3    = (const float*)d_in[16];
  const float* lWih   = (const float*)d_in[17];
  const float* lWhh   = (const float*)d_in[18];
  const float* lbih   = (const float*)d_in[19];
  const float* lbhh   = (const float*)d_in[20];
  const float* gate_W = (const float*)d_in[21];
  const float* gate_b = (const float*)d_in[22];
  const float* mW1    = (const float*)d_in[23];
  const float* mb1    = (const float*)d_in[24];
  const float* mW2    = (const float*)d_in[25];
  const float* mb2    = (const float*)d_in[26];
  const float* mW3    = (const float*)d_in[27];
  const float* mb3    = (const float*)d_in[28];

  const int NV = in_sizes[0];          // 100000
  const int NC = in_sizes[1];          // 400000
  const int E  = in_sizes[2];          // 600000

  // ---- workspace carve ----
  size_t off = 0;
  auto carve = [&](size_t bytes) {
    void* p = (char*)d_ws + off;
    off += (bytes + 255) & ~(size_t)255;
    return p;
  };
  float* msg   = (float*)carve((size_t)NC * H * 4);
  float* agg   = (float*)carve((size_t)NC * H * 4);
  float* h_v   = (float*)carve((size_t)NV * H * 4);
  float* c_v   = (float*)carve((size_t)NV * H * 4);
  float* h_c   = (float*)carve((size_t)NC * H * 4);
  float* c_c   = (float*)carve((size_t)NC * H * 4);
  unsigned short* eW1b = (unsigned short*)carve((size_t)4 * H * H * 2);
  unsigned short* eW2b = (unsigned short*)carve((size_t)4 * H * H * 2);
  unsigned short* eW3b = (unsigned short*)carve((size_t)4 * H * H * 2);
  unsigned short* Wihb = (unsigned short*)carve((size_t)2 * 4 * H * H * 2);
  unsigned short* Whhb = (unsigned short*)carve((size_t)2 * 4 * H * H * 2);
  float* logits = (float*)carve((size_t)NC * 4);
  float* pacc   = (float*)carve((size_t)(H + 2) * 4);  // [0..127]=acc, [128]=sumexp, [129]=maxenc
  float* psum   = pacc + H;
  unsigned int* pmax = (unsigned int*)(pacc + H + 1);

  // ---- weight bf16 conversion ----
  int nW = 4 * H * H;
  f2bf_kernel<<<(nW + 255) / 256, 256, 0, stream>>>(eW1, eW1b, nW);
  f2bf_kernel<<<(nW + 255) / 256, 256, 0, stream>>>(eW2, eW2b, nW);
  f2bf_kernel<<<(nW + 255) / 256, 256, 0, stream>>>(eW3, eW3b, nW);
  int nL = 2 * 4 * H * H;
  f2bf_kernel<<<(nL + 255) / 256, 256, 0, stream>>>(lWih, Wihb, nL);
  f2bf_kernel<<<(nL + 255) / 256, 256, 0, stream>>>(lWhh, Whhb, nL);

  // ---- init node states ----
  init_nodes_kernel<<<(int)(((size_t)NV * H + 255) / 256), 256, 0, stream>>>(var_x, embed, h_v, c_v, NV);
  init_nodes_kernel<<<(int)(((size_t)NC * H + 255) / 256), 256, 0, stream>>>(clause_x, embed, h_c, c_c, NC);

  const int gV = (NV + 31) / 32, gC = (NC + 31) / 32;
  const int gE = (E * 32 + 255) / 256;

  for (int step = 0; step < STEPS; ++step) {
    // relation 0: var -> clause  (etypes 0=pos, 1=neg), messages computed from c_v
    mlp3_kernel<<<gV, 256, 0, stream>>>(c_v, NV, eW1b, eb1, eW2b, eb2, eW3b, eb3, msg);
    hipMemsetAsync(agg, 0, (size_t)NC * H * 4, stream);
    scatter_kernel<<<gE, 256, 0, stream>>>(msg, pos_src, pos_dst, E, agg);
    mlp3_kernel<<<gV, 256, 0, stream>>>(c_v, NV, eW1b + 1 * H * H, eb1 + 1 * H,
                                        eW2b + 1 * H * H, eb2 + 1 * H,
                                        eW3b + 1 * H * H, eb3 + 1 * H, msg);
    scatter_kernel<<<gE, 256, 0, stream>>>(msg, neg_src, neg_dst, E, agg);
    lstm_kernel<<<gC, 256, 0, stream>>>(agg, h_c, c_c, Wihb, Whhb, lbih, lbhh, NC);

    // relation 1: clause -> var (etypes 2=pos_r, 3=neg_r), messages from c_c
    mlp3_kernel<<<gC, 256, 0, stream>>>(c_c, NC, eW1b + 2 * H * H, eb1 + 2 * H,
                                        eW2b + 2 * H * H, eb2 + 2 * H,
                                        eW3b + 2 * H * H, eb3 + 2 * H, msg);
    hipMemsetAsync(agg, 0, (size_t)NV * H * 4, stream);
    scatter_kernel<<<gE, 256, 0, stream>>>(msg, posr_src, posr_dst, E, agg);
    mlp3_kernel<<<gC, 256, 0, stream>>>(c_c, NC, eW1b + 3 * H * H, eb1 + 3 * H,
                                        eW2b + 3 * H * H, eb2 + 3 * H,
                                        eW3b + 3 * H * H, eb3 + 3 * H, msg);
    scatter_kernel<<<gE, 256, 0, stream>>>(msg, negr_src, negr_dst, E, agg);
    lstm_kernel<<<gV, 256, 0, stream>>>(agg, h_v, c_v, Wihb + 4 * H * H, Whhb + 4 * H * H,
                                        lbih + 4 * H, lbhh + 4 * H, NV);
  }

  // ---- global attention pooling over clause cell states + head MLP ----
  hipMemsetAsync(pacc, 0, (size_t)(H + 2) * 4, stream);
  pool_logits_kernel<<<NC, 128, 0, stream>>>(c_c, gate_W, gate_b, logits, pmax, NC);
  pool_accum_kernel<<<(NC + 127) / 128, 128, 0, stream>>>(c_c, logits, pmax, pacc, psum, NC);
  head_kernel<<<1, 128, 0, stream>>>(pacc, psum, mW1, mb1, mW2, mb2, mW3, mb3, (float*)d_out);
}
